// MambaLayer_temp_21466246545713
// MI455X (gfx1250) — compile-verified, matrix operand order checked by reading, not run on hardware
//
#include <hip/hip_runtime.h>
#include <hip/hip_bf16.h>
#include <math.h>

// ---------------- constants (reference shapes) ----------------
#define NB     4          // batch
#define DM     128        // d_model
#define DI     256        // d_inner
#define LSEQ   16384      // 128*128 tokens per batch item
#define MTOK   (NB*LSEQ)  // 65536 total tokens
#define DTRANK 8
#define DSTATE 16
#define DBLW   (DTRANK + 2*DSTATE)   // 40

typedef __attribute__((ext_vector_type(16))) __bf16 v16bf;
typedef __attribute__((ext_vector_type(8)))  __bf16 v8bf;
typedef __attribute__((ext_vector_type(8)))  float  v8f;
typedef __attribute__((ext_vector_type(4)))  unsigned v4u;
typedef __attribute__((ext_vector_type(8)))  int      v8i;
typedef __attribute__((ext_vector_type(4)))  int      v4i;

static __device__ __forceinline__ __bf16 f2bf(float f) {
  unsigned u = __builtin_bit_cast(unsigned, f);
  u += 0x7FFFu + ((u >> 16) & 1u);          // round-to-nearest-even
  unsigned short h = (unsigned short)(u >> 16);
  return __builtin_bit_cast(__bf16, h);
}

// ---------------- 1) pack weights f32 -> bf16 in WMMA B-fragment order ----------
// Packed layout: P[((tc*KB + kb32)*512) + lane*16 + j]  (tc = n-tile, kb32 = k-block)
// lane's 16 elements j: k = kb + j (j<8) | kb + 16 + (j-8), kb = (lane>>4)*8; col = tc*16 + (lane&15)
template <int K, int N>
__global__ __launch_bounds__(256)
void k_pack_w(const float* __restrict__ W, __bf16* __restrict__ P) {
  constexpr int KB = K / 32;
  const int idx  = blockIdx.x * 256 + threadIdx.x;     // 0 .. N*K-1
  const int j    = idx & 15;
  const int lane = (idx >> 4) & 31;
  const int rest = idx >> 9;
  const int kb32 = rest % KB;
  const int tc   = rest / KB;
  const int kb   = (lane >> 4) * 8;
  const int kidx = (j < 8) ? (kb + j) : (kb + 16 + (j - 8));
  const int k    = kb32 * 32 + kidx;
  const int col  = tc * 16 + (lane & 15);
  P[idx] = f2bf(W[(size_t)k * N + col]);
}

// ---------------- 2) layernorm over contiguous feature rows -> bf16 ----------------
__global__ __launch_bounds__(256)
void k_layernorm(const float* __restrict__ x, const float* __restrict__ gamma,
                 const float* __restrict__ beta, __bf16* __restrict__ xn) {
  const int warp = threadIdx.x >> 5, lane = threadIdx.x & 31;
  const int tok  = blockIdx.x * 8 + warp;         // one wave per token
  const float4 v = reinterpret_cast<const float4*>(x + (size_t)tok * DM)[lane];
  float s  = v.x + v.y + v.z + v.w;
  float sq = v.x*v.x + v.y*v.y + v.z*v.z + v.w*v.w;
  #pragma unroll
  for (int off = 16; off > 0; off >>= 1) {
    s  += __shfl_xor(s,  off, 32);
    sq += __shfl_xor(sq, off, 32);
  }
  const float mean = s * (1.f / DM);
  const float var  = sq * (1.f / DM) - mean * mean;
  const float rstd = rsqrtf(var + 1e-5f);
  const int c0 = lane * 4;
  const float vv[4] = {v.x, v.y, v.z, v.w};
  #pragma unroll
  for (int j = 0; j < 4; ++j)
    xn[(size_t)tok * DM + c0 + j] = f2bf((vv[j] - mean) * rstd * gamma[c0 + j] + beta[c0 + j]);
}

// ---------------- 3/8) WMMA GEMM with TDM-staged packed weights -------------------
// C(MxN) = A(MxK) @ W(KxN). Wave computes 32 rows x 64 cols (MT=2 x NT=4 tiles):
// each B-fragment (from LDS) is reused by 2 A-fragments; each A-fragment feeds 4 WMMAs.
// All 8 waves of a block share one n-group; its packed weight panel (one contiguous
// 1-D span, K*128 bytes) is DMA'd to LDS once per block by the Tensor Data Mover.
template <int K, int N, bool SCATTER>
__global__ __launch_bounds__(256)
void k_gemm_bf16(const __bf16* __restrict__ A, const __bf16* __restrict__ Bp,
                 float* __restrict__ C) {
  constexpr int MT = 2;                 // m-tiles per wave
  constexpr int NT = 4;                 // n-tiles per wave
  constexpr int KB = K / 32;            // k-blocks
  constexpr int BPC = (MTOK / (16 * MT)) / 8;  // blocks per n-group column (256)
  __shared__ __bf16 smemB[NT * KB * 512];      // K*64 bf16 = K*128 bytes

  const int lane = threadIdx.x & 31;
  const int warp = threadIdx.x >> 5;
  const int ng   = blockIdx.x / BPC;                     // uniform per block
  const int tmw  = (blockIdx.x % BPC) * 8 + warp;        // wave's 32-row strip
  const int half = lane >> 4;
  const int mn   = lane & 15;
  const int kboff = half * 8;

  // ---- stage this n-group's packed panel into LDS via TDM (one DMA per block) ----
  if (warp == 0) {
    const unsigned long long ga =
        (unsigned long long)(uintptr_t)(Bp + (size_t)ng * (NT * KB * 512));
    const unsigned lds = (unsigned)(uintptr_t)&smemB[0];
    constexpr unsigned NELEM = (unsigned)K * 32u;        // panel size in dwords
    v4u g0;
    g0.x = 1u;                                           // count = 1 valid descriptor
    g0.y = lds;                                          // lds_addr
    g0.z = (unsigned)ga;                                 // global_addr[31:0]
    g0.w = (unsigned)((ga >> 32) & 0x01FFFFFFull) | (2u << 30);  // addr[56:32] | type=2
    v8i g1;
    g1[0] = (int)(2u << 16);                  // wg_mask=0 (no cluster), data_size=4B
    g1[1] = (int)((NELEM & 0xFFFFu) << 16);   // tensor_dim0[15:0]
    g1[2] = (int)(1u << 16);                  // tensor_dim0 hi=0, tensor_dim1=1
    g1[3] = (int)(NELEM << 16);               // tile_dim0 = whole panel (1-D)
    g1[4] = 1;                                // tile_dim1 = 1
    g1[5] = (int)NELEM;                       // tensor_dim0_stride
    g1[6] = 0;
    g1[7] = 0;
    v4i gz = {0, 0, 0, 0};
#if defined(__clang_major__) && (__clang_major__ >= 23)
    v8i gz8 = {0, 0, 0, 0, 0, 0, 0, 0};
    __builtin_amdgcn_tensor_load_to_lds(g0, g1, gz, gz, gz8, 0);
#else
    __builtin_amdgcn_tensor_load_to_lds(g0, g1, gz, gz, 0);
#endif
  }
  __builtin_amdgcn_s_wait_tensorcnt((short)0);   // issuer drains TENSORcnt; others no-op
  __syncthreads();

  // ---- main loop ----
  v8f acc[MT][NT] = {};
  const __bf16* arow0 = A + (size_t)(tmw * 32 + mn) * K;       // m-tile 0
  const __bf16* arow1 = arow0 + (size_t)16 * K;                // m-tile 1
  #pragma unroll
  for (int kb32 = 0; kb32 < KB; ++kb32) {
    // A fragments: 2x (two b128 loads, contiguous runs at +0 / +32B)
    const __bf16* ap0 = arow0 + kb32 * 32 + kboff;
    const __bf16* ap1 = arow1 + kb32 * 32 + kboff;
    const v16bf a0 = __builtin_shufflevector(
        *reinterpret_cast<const v8bf*>(ap0), *reinterpret_cast<const v8bf*>(ap0 + 16),
        0, 1, 2, 3, 4, 5, 6, 7, 8, 9, 10, 11, 12, 13, 14, 15);
    const v16bf a1 = __builtin_shufflevector(
        *reinterpret_cast<const v8bf*>(ap1), *reinterpret_cast<const v8bf*>(ap1 + 16),
        0, 1, 2, 3, 4, 5, 6, 7, 8, 9, 10, 11, 12, 13, 14, 15);
    // B fragments: batch all NT LDS loads first -> single ds clause, single wait
    v16bf b[NT];
    #pragma unroll
    for (int nt = 0; nt < NT; ++nt)
      b[nt] = *reinterpret_cast<const v16bf*>(
          &smemB[((nt * KB + kb32) << 9) + (lane << 4)]);
    // 8 WMMAs per k-step
    #pragma unroll
    for (int nt = 0; nt < NT; ++nt)
      acc[0][nt] = __builtin_amdgcn_wmma_f32_16x16x32_bf16(
          false, a0, false, b[nt], (short)0, acc[0][nt], false, false);
    #pragma unroll
    for (int nt = 0; nt < NT; ++nt)
      acc[1][nt] = __builtin_amdgcn_wmma_f32_16x16x32_bf16(
          false, a1, false, b[nt], (short)0, acc[1][nt], false, false);
  }

  // ---- epilogue ----
  #pragma unroll
  for (int mt = 0; mt < MT; ++mt) {
    #pragma unroll
    for (int nt = 0; nt < NT; ++nt) {
      const int col = (ng * NT + nt) * 16 + mn;
      #pragma unroll
      for (int v = 0; v < 8; ++v) {
        const int r = tmw * 32 + mt * 16 + half * 8 + v;
        if (SCATTER) {
          const int bI = r >> 14, l = r & (LSEQ - 1);
          C[(size_t)bI * (DM * DM * DM) + (size_t)col * LSEQ + l] = acc[mt][nt][v];
        } else {
          C[(size_t)r * N + col] = acc[mt][nt][v];
        }
      }
    }
  }
}

// ---------------- 4) depthwise causal conv (d_conv=4) + SiLU ----------------
__global__ __launch_bounds__(256)
void k_conv_silu(const float* __restrict__ xz, const float* __restrict__ convw,
                 const float* __restrict__ convb, float* __restrict__ u) {
  const int idx = blockIdx.x * 256 + threadIdx.x;  // MTOK*DI threads
  const int i = idx & (DI - 1);
  const int r = idx >> 8;
  const int l = r & (LSEQ - 1);
  float acc = convb[i];
  #pragma unroll
  for (int k = 0; k < 4; ++k) {
    const int lk = l - 3 + k;
    if (lk >= 0) acc += xz[(size_t)(r - 3 + k) * (2 * DI) + i] * convw[i * 4 + k];
  }
  u[idx] = acc / (1.f + __expf(-acc));             // silu
}

// ---------------- 5) x_proj: dbl(M x 40) = u(M x 256) @ W(256 x 40) ----------------
__global__ __launch_bounds__(256)
void k_xproj(const float* __restrict__ u, const float* __restrict__ xw,
             float* __restrict__ dbl) {
  const int idx = blockIdx.x * 256 + threadIdx.x;  // MTOK*40 threads
  const int j = idx % DBLW;
  const int r = idx / DBLW;
  const float* ur = u + (size_t)r * DI;
  float acc = 0.f;
  #pragma unroll 8
  for (int i = 0; i < DI; ++i) acc += ur[i] * xw[i * DBLW + j];
  dbl[idx] = acc;
}

// ---------------- 6) dt_proj + softplus ----------------
__global__ __launch_bounds__(256)
void k_dtproj(const float* __restrict__ dbl, const float* __restrict__ dtw,
              const float* __restrict__ dtb, float* __restrict__ dt) {
  const int idx = blockIdx.x * 256 + threadIdx.x;  // MTOK*DI threads
  const int i = idx & (DI - 1);
  const int r = idx >> 8;
  float acc = dtb[i];
  #pragma unroll
  for (int q = 0; q < DTRANK; ++q) acc += dbl[(size_t)r * DBLW + q] * dtw[q * DI + i];
  dt[idx] = (acc > 20.f) ? acc : log1pf(__expf(acc));
}

// ---------------- 7) selective scan, fused output gate, bf16 out ----------------
// thread = (b, i, n): n = tid&15, i = (tid>>4)&255, b = tid>>12. 16384 threads.
__global__ __launch_bounds__(256)
void k_scan(const float* __restrict__ dt, const float* __restrict__ u,
            const float* __restrict__ dbl, const float* __restrict__ xz,
            const float* __restrict__ A_log, const float* __restrict__ Dp,
            __bf16* __restrict__ ygate) {
  const int tid = blockIdx.x * 256 + threadIdx.x;
  const int n = tid & 15;
  const int i = (tid >> 4) & (DI - 1);
  const int b = tid >> 12;
  const float Ain = -__expf(A_log[i * DSTATE + n]);
  const float Dv  = Dp[i];
  float h = 0.f;
  const int base = b * LSEQ;
  for (int l = 0; l < LSEQ; ++l) {
    const int r = base + l;
    const float dtv = dt[(size_t)r * DI + i];
    const float uv  = u[(size_t)r * DI + i];
    const float Bv  = dbl[(size_t)r * DBLW + DTRANK + n];
    const float Cv  = dbl[(size_t)r * DBLW + DTRANK + DSTATE + n];
    h = h * __expf(dtv * Ain) + dtv * Bv * uv;
    float yv = h * Cv;
    yv += __shfl_xor(yv, 1, 32);                 // reduce over 16 states
    yv += __shfl_xor(yv, 2, 32);
    yv += __shfl_xor(yv, 4, 32);
    yv += __shfl_xor(yv, 8, 32);
    if (n == 0) {
      const float zv = xz[(size_t)r * (2 * DI) + DI + i];
      const float y  = yv + uv * Dv;
      const float g  = zv / (1.f + __expf(-zv)); // silu(z)
      ygate[(size_t)r * DI + i] = f2bf(y * g);
    }
  }
}

// ---------------- launcher ----------------
extern "C" void kernel_launch(void* const* d_in, const int* in_sizes, int n_in,
                              void* d_out, int out_size, void* d_ws, size_t ws_size,
                              hipStream_t stream) {
  const float* x        = (const float*)d_in[0];
  const float* gamma    = (const float*)d_in[1];
  const float* beta     = (const float*)d_in[2];
  const float* in_w     = (const float*)d_in[3];
  const float* conv_w   = (const float*)d_in[4];
  const float* conv_b   = (const float*)d_in[5];
  const float* x_proj_w = (const float*)d_in[6];
  const float* dt_w     = (const float*)d_in[7];
  const float* dt_b     = (const float*)d_in[8];
  const float* A_log    = (const float*)d_in[9];
  const float* Dp       = (const float*)d_in[10];
  const float* out_w    = (const float*)d_in[11];
  float* out = (float*)d_out;

  char* ws = (char*)d_ws;
  size_t off = 0;
  auto alloc = [&](size_t bytes) -> char* {
    char* p = ws + off;
    off = (off + bytes + 255) & ~(size_t)255;
    return p;
  };
  __bf16* winP  = (__bf16*)alloc((size_t)DM * 2 * DI * 2);      // packed 128x512 bf16
  __bf16* woutP = (__bf16*)alloc((size_t)DI * DM * 2);          // packed 256x128 bf16
  __bf16* xnB   = (__bf16*)alloc((size_t)MTOK * DM * 2);        // 16 MB
  float*  xz    = (float*)alloc((size_t)MTOK * 2 * DI * 4);     // 128 MB
  float*  u     = (float*)alloc((size_t)MTOK * DI * 4);         // 64 MB
  float*  dbl   = (float*)alloc((size_t)MTOK * DBLW * 4);       // 10 MB
  float*  dt    = (float*)alloc((size_t)MTOK * DI * 4);         // 64 MB
  __bf16* ygate = (__bf16*)alloc((size_t)MTOK * DI * 2);        // 32 MB
  (void)ws_size; (void)in_sizes; (void)n_in; (void)out_size;

  // 1) pack weights into WMMA fragment order (bf16)
  k_pack_w<DM, 2 * DI><<<(DM * 2 * DI) / 256, 256, 0, stream>>>(in_w, winP);
  k_pack_w<DI, DM><<<(DI * DM) / 256, 256, 0, stream>>>(out_w, woutP);
  // 2) layernorm (token feature vectors are contiguous in mode-2 layout)
  k_layernorm<<<MTOK / 8, 256, 0, stream>>>(x, gamma, beta, xnB);
  // 3) in_proj GEMM: 65536x128 @ 128x512 -> xz  (WMMA bf16, TDM-staged weights)
  k_gemm_bf16<DM, 2 * DI, false>
      <<<(2 * DI / 64) * ((MTOK / 32) / 8), 256, 0, stream>>>(xnB, winP, xz);
  // 4) depthwise causal conv + silu -> u
  k_conv_silu<<<(MTOK * DI) / 256, 256, 0, stream>>>(xz, conv_w, conv_b, u);
  // 5) x_proj -> dbl (dt_low | B | C)
  k_xproj<<<(MTOK * DBLW) / 256, 256, 0, stream>>>(u, x_proj_w, dbl);
  // 6) dt_proj + softplus -> dt
  k_dtproj<<<(MTOK * DI) / 256, 256, 0, stream>>>(dbl, dt_w, dt_b, dt);
  // 7) selective scan + skip + gate -> ygate (bf16)
  k_scan<<<(NB * DI * DSTATE) / 256, 256, 0, stream>>>(dt, u, dbl, xz, A_log, Dp, ygate);
  // 8) out_proj GEMM: 65536x256 @ 256x128 -> scattered transposed f32 output
  k_gemm_bf16<DI, DM, true>
      <<<(DM / 64) * ((MTOK / 32) / 8), 256, 0, stream>>>(ygate, woutP, out);
}
